// RelationalGraphBackbone_42262478192901
// MI455X (gfx1250) — matrix operand order repaired and checked by reading, not compiled, not run on hardware
//
#include <hip/hip_runtime.h>

// ---------------------------------------------------------------------------
// RelationalGraphBackbone on MI455X (gfx1250): bf16 WMMA GEMMs + edge softmax
//   - weights pre-swizzled once into WMMA B-fragment order (bf16)
//   - user rows padded to a multiple of 32 -> branch-free 2-tile GEMM waves
//   - B fragments preloaded per column tile (load clause, then 8 WMMAs)
// ---------------------------------------------------------------------------

#define N_USER   50000
#define N_PADU   50016      // next multiple of 32
#define N_TWEET  100000     // already multiple of 32
#define N_EDGE   100000
#define DIM      128
#define HCDIM    512
#define NHEAD    4

typedef __attribute__((ext_vector_type(16))) __bf16 v16bf;
typedef __attribute__((ext_vector_type(8)))  __bf16 v8bf;
typedef __attribute__((ext_vector_type(8)))  float  v8f;

// ---- bf16 helpers (manual bit manipulation; RNE rounding) ------------------
__device__ __forceinline__ unsigned short f2bf_u16(float f) {
    unsigned int u = __builtin_bit_cast(unsigned int, f);
    u += 0x7FFFu + ((u >> 16) & 1u);
    return (unsigned short)(u >> 16);
}
__device__ __forceinline__ float bf2f(unsigned short h) {
    unsigned int u = ((unsigned int)h) << 16;
    return __builtin_bit_cast(float, u);
}

// ---- order-preserving float<->uint encoding for atomic segment-max ---------
__device__ __forceinline__ unsigned int encF(float x) {
    unsigned int u = __builtin_bit_cast(unsigned int, x);
    return (u & 0x80000000u) ? ~u : (u | 0x80000000u);
}
__device__ __forceinline__ float decF(unsigned int u) {
    u = (u & 0x80000000u) ? (u & 0x7FFFFFFFu) : ~u;
    return __builtin_bit_cast(float, u);
}

// ---------------------------------------------------------------------------
// Weight swizzle: W f32 [nMat][128, NC] -> bf16 B-fragment order:
//   out[ mat ][ ((ct*4 + kt)*32 + lane)*16 + j ] =
//       bf16( W[mat][ (kt*32 + (lane>>4)*16 + j) * NC + ct*16 + (lane&15) ] )
// so each wave-lane's 16-element B fragment is one contiguous 32B chunk.
// ---------------------------------------------------------------------------
__global__ void __launch_bounds__(256)
swizzle_weights(const float* __restrict__ W, unsigned short* __restrict__ out,
                int NC, int total)
{
    const int idx = blockIdx.x * 256 + threadIdx.x;
    if (idx >= total) return;
    const int per  = 128 * NC;
    const int mat  = idx / per;
    const int r    = idx - mat * per;
    const int j    = r & 15;
    const int lane = (r >> 4) & 31;
    const int kt   = (r >> 9) & 3;
    const int ct   = r >> 11;
    const int k = kt * 32 + (lane >> 4) * 16 + j;
    const int n = ct * 16 + (lane & 15);
    out[idx] = f2bf_u16(W[(size_t)mat * per + (size_t)k * NC + n]);
}

// ---------------------------------------------------------------------------
// WMMA GEMM: Y[N, NC] = Xbf16[N,128] @ Wsw(bf16, fragment order) + bias
//   NC = 512 (q/k/v proj, bf16 out) or 128 (skip, accumulate f32 out += ...)
// One wave per PAIR of 16-row tiles (N is a multiple of 32; no ragged tiles).
// ---------------------------------------------------------------------------
template<int NC, bool ACCUM>
__global__ void __launch_bounds__(256)
gemm_bf16_wmma(const unsigned short* __restrict__ Xb,
               const unsigned short* __restrict__ Wsw,
               const float* __restrict__ bias,
               void* __restrict__ Yout,
               int nPairs)
{
    const int lane = threadIdx.x & 31;
    const int wid  = threadIdx.x >> 5;
    const int pair = blockIdx.x * 8 + wid;
    if (pair >= nPairs) return;             // wave-uniform: branch, no partial EXEC
    const int row0 = pair * 32;

    const int m    = lane & 15;
    const int half = lane >> 4;             // 0 or 1

    // A fragments (16x32 bf16), contiguous k order, same as swizzled B:
    //  lanes 0-15 : K = {0..15}+32*kt ; lanes 16-31: K = {16..31}+32*kt
    v16bf afrag0[4], afrag1[4];
    const unsigned short* xr0 = Xb + (size_t)(row0 + m) * DIM + half * 16;
    const unsigned short* xr1 = xr0 + 16 * DIM;
    #pragma unroll
    for (int kt = 0; kt < 4; ++kt) {
        v8bf lo0 = *(const v8bf*)(xr0 + kt * 32);
        v8bf hi0 = *(const v8bf*)(xr0 + kt * 32 + 8);
        afrag0[kt] = __builtin_shufflevector(lo0, hi0,
            0,1,2,3,4,5,6,7,8,9,10,11,12,13,14,15);
        v8bf lo1 = *(const v8bf*)(xr1 + kt * 32);
        v8bf hi1 = *(const v8bf*)(xr1 + kt * 32 + 8);
        afrag1[kt] = __builtin_shufflevector(lo1, hi1,
            0,1,2,3,4,5,6,7,8,9,10,11,12,13,14,15);
    }

    const int n = lane & 15;

    #pragma unroll 1
    for (int ct = 0; ct < NC / 16; ++ct) {
        const int cbase = ct * 16;
        // preload all 4 B fragments (contiguous 32B per lane; 8 b128 loads)
        v16bf b[4];
        #pragma unroll
        for (int kt = 0; kt < 4; ++kt)
            b[kt] = *(const v16bf*)(
                Wsw + (((size_t)(ct * 4 + kt) * 32 + lane) << 4));
        v8f acc0 = {}, acc1 = {};
        #pragma unroll
        for (int kt = 0; kt < 4; ++kt) {
            acc0 = __builtin_amdgcn_wmma_f32_16x16x32_bf16(
                false, afrag0[kt], false, b[kt], (short)0, acc0, false, false);
            acc1 = __builtin_amdgcn_wmma_f32_16x16x32_bf16(
                false, afrag1[kt], false, b[kt], (short)0, acc1, false, false);
        }
        const float bv = bias[cbase + n];
        if (ACCUM) {
            float* Y = (float*)Yout;
            #pragma unroll
            for (int r = 0; r < 8; ++r) {
                const int mr = r + half * 8;    // C/D layout: M = r + 8*half
                Y[(size_t)(row0 + mr) * NC + cbase + n] += acc0[r] + bv;
            }
            #pragma unroll
            for (int r = 0; r < 8; ++r) {
                const int mr = r + half * 8;
                Y[(size_t)(row0 + 16 + mr) * NC + cbase + n] += acc1[r] + bv;
            }
        } else {
            unsigned short* Y = (unsigned short*)Yout;
            #pragma unroll
            for (int r = 0; r < 8; ++r) {
                const int mr = r + half * 8;
                Y[(size_t)(row0 + mr) * NC + cbase + n] = f2bf_u16(acc0[r] + bv);
            }
            #pragma unroll
            for (int r = 0; r < 8; ++r) {
                const int mr = r + half * 8;
                Y[(size_t)(row0 + 16 + mr) * NC + cbase + n] =
                    f2bf_u16(acc1[r] + bv);
            }
        }
    }
}

// ---------------------------------------------------------------------------
// Edge kernels
// ---------------------------------------------------------------------------
// wave per edge: logits[e,h] = (q[dst,h,:].k[src,h,:]) / sqrt(128); seg-max
__global__ void __launch_bounds__(256)
edge_logits(const int* __restrict__ edges,            // [2,E]: src row, dst row
            const unsigned short* __restrict__ Qb,    // [n_dst,512] bf16
            const unsigned short* __restrict__ Kb,    // [n_src,512] bf16
            float* __restrict__ logits,               // [E,4]
            unsigned int* __restrict__ m_enc,         // [n_dst,4] encoded max
            int nE)
{
    const int e = blockIdx.x * 8 + (threadIdx.x >> 5);
    if (e >= nE) return;
    const int lane = threadIdx.x & 31;
    const int src = edges[e];
    const int dst = edges[nE + e];
    const unsigned short* qp = Qb + (size_t)dst * HCDIM + lane * 16;
    const unsigned short* kp = Kb + (size_t)src * HCDIM + lane * 16;
    float s = 0.f;
    #pragma unroll
    for (int j = 0; j < 16; ++j) s += bf2f(qp[j]) * bf2f(kp[j]);
    s += __shfl_xor(s, 1);
    s += __shfl_xor(s, 2);
    s += __shfl_xor(s, 4);                  // reduce within 8-lane head group
    if ((lane & 7) == 0) {
        const int h = lane >> 3;
        const float lg = s * 0.08838834764831845f;   // 1/sqrt(128)
        logits[(size_t)e * NHEAD + h] = lg;
        atomicMax(&m_enc[(size_t)dst * NHEAD + h], encF(lg));
    }
}

// thread per (edge,head): e = exp(logit-m[dst]); z[dst,h] += e (in-place buf)
__global__ void __launch_bounds__(256)
edge_exp(const int* __restrict__ edges,
         float* __restrict__ logits,
         const unsigned int* __restrict__ m_enc,
         float* __restrict__ z, int nE)
{
    const int t = blockIdx.x * 256 + threadIdx.x;
    if (t >= nE * NHEAD) return;
    const int e = t >> 2, h = t & 3;
    const int dst = edges[nE + e];
    const float mv = decF(m_enc[(size_t)dst * NHEAD + h]);
    const float ev = __expf(logits[t] - mv);
    logits[t] = ev;
    unsafeAtomicAdd(&z[(size_t)dst * NHEAD + h], ev);
}

// wave per edge: agg[dst,d] += (1/4) * sum_h (e/z) * v[src,h,d]
__global__ void __launch_bounds__(256)
edge_scatter(const int* __restrict__ edges,
             const float* __restrict__ ebuf,          // [E,4] exp values
             const float* __restrict__ z,             // [n_dst,4]
             const unsigned short* __restrict__ Vb,   // [n_src,512] bf16
             float* __restrict__ agg, int nE)
{
    const int e = blockIdx.x * 8 + (threadIdx.x >> 5);
    if (e >= nE) return;
    const int lane = threadIdx.x & 31;
    const int src = edges[e];
    const int dst = edges[nE + e];
    const float a0 = ebuf[(size_t)e * 4 + 0] / z[(size_t)dst * 4 + 0];
    const float a1 = ebuf[(size_t)e * 4 + 1] / z[(size_t)dst * 4 + 1];
    const float a2 = ebuf[(size_t)e * 4 + 2] / z[(size_t)dst * 4 + 2];
    const float a3 = ebuf[(size_t)e * 4 + 3] / z[(size_t)dst * 4 + 3];
    const unsigned short* vp = Vb + (size_t)src * HCDIM;
    const int d0 = lane * 4;
    #pragma unroll
    for (int j = 0; j < 4; ++j) {
        const int d = d0 + j;
        const float val = a0 * bf2f(vp[d])       + a1 * bf2f(vp[128 + d])
                        + a2 * bf2f(vp[256 + d]) + a3 * bf2f(vp[384 + d]);
        unsafeAtomicAdd(&agg[(size_t)dst * DIM + d], 0.25f * val);
    }
}

// ---------------------------------------------------------------------------
// LayerNorm(agg)*g + b + x -> x (in place), wave per node
// ---------------------------------------------------------------------------
__global__ void __launch_bounds__(256)
layernorm_residual(const float* __restrict__ agg,
                   const float* __restrict__ g, const float* __restrict__ b,
                   float* __restrict__ x, int nNodes)
{
    const int node = blockIdx.x * 8 + (threadIdx.x >> 5);
    if (node >= nNodes) return;
    const int lane = threadIdx.x & 31;
    const float* ap = agg + (size_t)node * DIM;
    float v[4];
    #pragma unroll
    for (int j = 0; j < 4; ++j) v[j] = ap[lane + 32 * j];
    float s = v[0] + v[1] + v[2] + v[3];
    #pragma unroll
    for (int o = 1; o < 32; o <<= 1) s += __shfl_xor(s, o);
    const float mean = s * (1.0f / DIM);
    float q = 0.f;
    #pragma unroll
    for (int j = 0; j < 4; ++j) { const float d = v[j] - mean; q += d * d; }
    #pragma unroll
    for (int o = 1; o < 32; o <<= 1) q += __shfl_xor(q, o);
    const float inv = rsqrtf(q * (1.0f / DIM) + 1e-5f);
    #pragma unroll
    for (int j = 0; j < 4; ++j) {
        const int d = lane + 32 * j;
        const size_t idx = (size_t)node * DIM + d;
        x[idx] = g[d] * (v[j] - mean) * inv + b[d] + x[idx];
    }
}

// ---------------------------------------------------------------------------
// Small utility kernels
// ---------------------------------------------------------------------------
// convert first nValid elements, zero-fill up to nTotal (row padding)
__global__ void __launch_bounds__(256)
f32_to_bf16_pad(const float* __restrict__ in, unsigned short* __restrict__ out,
                int nValid, int nTotal)
{
    const int i = blockIdx.x * 256 + threadIdx.x;
    if (i >= nTotal) return;
    out[i] = (i < nValid) ? f2bf_u16(in[i]) : (unsigned short)0;
}

__global__ void __launch_bounds__(256)
bcast_tweet_k(const float* __restrict__ tv, float* __restrict__ xt, int n)
{
    const int i = blockIdx.x * 256 + threadIdx.x;
    if (i < n) xt[i] = tv[i & (DIM - 1)];
}

// ---------------------------------------------------------------------------
// Host orchestration
// ---------------------------------------------------------------------------
extern "C" void kernel_launch(void* const* d_in, const int* in_sizes, int n_in,
                              void* d_out, int out_size, void* d_ws, size_t ws_size,
                              hipStream_t stream)
{
    const float* emb_user = (const float*)d_in[0];
    const float* tv_tweet = (const float*)d_in[1];
    const float* W_qkv    = (const float*)d_in[2];   // [2,3,3,128,512]
    const float* b_qkv    = (const float*)d_in[3];   // [2,3,3,512]
    const float* W_skip   = (const float*)d_in[4];   // [2,3,128,128]
    const float* b_skip   = (const float*)d_in[5];   // [2,3,128]
    const float* ln_g     = (const float*)d_in[6];   // [2,2,128]
    const float* ln_b     = (const float*)d_in[7];   // [2,2,128]
    const int*   edge[3]  = {(const int*)d_in[8],    // uu [2,E]
                             (const int*)d_in[9],    // ut
                             (const int*)d_in[10]};  // tu

    // ---- workspace carve-up (all chunks are 256B multiples) ----
    char* ws = (char*)d_ws;
    size_t off = 0;
    auto carve = [&](size_t bytes) -> char* {
        char* p = ws + off;
        off += (bytes + 255) & ~(size_t)255;
        return p;
    };
    float*          x_user  = (float*)         carve((size_t)N_USER  * DIM * 4);
    float*          x_tweet = (float*)         carve((size_t)N_TWEET * DIM * 4);
    unsigned short* xu_bf   = (unsigned short*)carve((size_t)N_PADU  * DIM * 2);
    unsigned short* xt_bf   = (unsigned short*)carve((size_t)N_TWEET * DIM * 2);
    unsigned short* q_bf    = (unsigned short*)carve((size_t)N_TWEET * HCDIM * 2);
    unsigned short* k_bf    = (unsigned short*)carve((size_t)N_TWEET * HCDIM * 2);
    unsigned short* v_bf    = (unsigned short*)carve((size_t)N_TWEET * HCDIM * 2);
    float*          ebuf    = (float*)         carve((size_t)N_EDGE  * NHEAD * 4);
    unsigned int*   m_enc   = (unsigned int*)  carve((size_t)N_TWEET * NHEAD * 4);
    float*          zbuf    = (float*)         carve((size_t)N_TWEET * NHEAD * 4);
    float*          agg_u   = (float*)         carve((size_t)N_PADU  * DIM * 4);
    float*          agg_t   = (float*)         carve((size_t)N_TWEET * DIM * 4);
    unsigned short* wqkv_sw = (unsigned short*)carve((size_t)18 * DIM * HCDIM * 2);
    unsigned short* wskp_sw = (unsigned short*)carve((size_t)6  * DIM * DIM * 2);
    (void)ws_size; (void)in_sizes; (void)n_in;

    const int srcT[3] = {0, 0, 1};   // 0=user, 1=tweet
    const int dstT[3] = {0, 1, 0};
    const int nNodes[2]  = {N_USER, N_TWEET};   // logical counts (edges, LN)
    const int nPadded[2] = {N_PADU, N_TWEET};   // GEMM row counts (mult. of 32)

    // ---- one-time (per launch) weight swizzle to bf16 fragment order ----
    {
        const int totQ = 18 * DIM * HCDIM;   // 1,179,648
        const int totS = 6 * DIM * DIM;      // 98,304
        swizzle_weights<<<(totQ + 255) / 256, 256, 0, stream>>>(
            W_qkv, wqkv_sw, HCDIM, totQ);
        swizzle_weights<<<(totS + 255) / 256, 256, 0, stream>>>(
            W_skip, wskp_sw, DIM, totS);
    }

    // ---- init node states ----
    hipMemcpyAsync(x_user, emb_user, (size_t)N_USER * DIM * 4,
                   hipMemcpyDeviceToDevice, stream);
    bcast_tweet_k<<<(N_TWEET * DIM + 255) / 256, 256, 0, stream>>>(
        tv_tweet, x_tweet, N_TWEET * DIM);

    for (int l = 0; l < 2; ++l) {
        f32_to_bf16_pad<<<(N_PADU * DIM + 255) / 256, 256, 0, stream>>>(
            x_user, xu_bf, N_USER * DIM, N_PADU * DIM);
        f32_to_bf16_pad<<<(N_TWEET * DIM + 255) / 256, 256, 0, stream>>>(
            x_tweet, xt_bf, N_TWEET * DIM, N_TWEET * DIM);
        hipMemsetAsync(agg_u, 0, (size_t)N_PADU * DIM * 4, stream);
        hipMemsetAsync(agg_t, 0, (size_t)N_TWEET * DIM * 4, stream);

        for (int r = 0; r < 3; ++r) {
            const unsigned short* xsrc = srcT[r] ? xt_bf : xu_bf;
            const unsigned short* xdst = dstT[r] ? xt_bf : xu_bf;
            const int nD = nNodes[dstT[r]];
            float* agg = dstT[r] ? agg_t : agg_u;
            const int pairsD = nPadded[dstT[r]] / 32;
            const int pairsS = nPadded[srcT[r]] / 32;

            // skip: agg += x_dst @ Ws + bs
            const unsigned short* Ws = wskp_sw + ((size_t)l * 3 + r) * DIM * DIM;
            const float*          bs = b_skip + ((size_t)l * 3 + r) * DIM;
            gemm_bf16_wmma<DIM, true><<<(pairsD + 7) / 8, 256, 0, stream>>>(
                xdst, Ws, bs, agg, pairsD);

            // q (dst), k (src), v (src) projections -> bf16 [n,512]
            const size_t wb = (((size_t)l * 3 + r) * 3);
            const unsigned short* Wq = wqkv_sw + (wb + 0) * DIM * HCDIM;
            const unsigned short* Wk = wqkv_sw + (wb + 1) * DIM * HCDIM;
            const unsigned short* Wv = wqkv_sw + (wb + 2) * DIM * HCDIM;
            const float* bq = b_qkv + (wb + 0) * HCDIM;
            const float* bk = b_qkv + (wb + 1) * HCDIM;
            const float* bv = b_qkv + (wb + 2) * HCDIM;
            gemm_bf16_wmma<HCDIM, false><<<(pairsD + 7) / 8, 256, 0, stream>>>(
                xdst, Wq, bq, q_bf, pairsD);
            gemm_bf16_wmma<HCDIM, false><<<(pairsS + 7) / 8, 256, 0, stream>>>(
                xsrc, Wk, bk, k_bf, pairsS);
            gemm_bf16_wmma<HCDIM, false><<<(pairsS + 7) / 8, 256, 0, stream>>>(
                xsrc, Wv, bv, v_bf, pairsS);

            // segment softmax + scatter
            hipMemsetAsync(m_enc, 0, (size_t)nD * NHEAD * 4, stream);
            hipMemsetAsync(zbuf,  0, (size_t)nD * NHEAD * 4, stream);
            edge_logits<<<(N_EDGE + 7) / 8, 256, 0, stream>>>(
                edge[r], q_bf, k_bf, ebuf, m_enc, N_EDGE);
            edge_exp<<<(N_EDGE * NHEAD + 255) / 256, 256, 0, stream>>>(
                edge[r], ebuf, m_enc, zbuf, N_EDGE);
            edge_scatter<<<(N_EDGE + 7) / 8, 256, 0, stream>>>(
                edge[r], ebuf, zbuf, v_bf, agg, N_EDGE);
        }

        layernorm_residual<<<(N_USER + 7) / 8, 256, 0, stream>>>(
            agg_u, ln_g + ((size_t)l * 2 + 0) * DIM, ln_b + ((size_t)l * 2 + 0) * DIM,
            x_user, N_USER);
        layernorm_residual<<<(N_TWEET + 7) / 8, 256, 0, stream>>>(
            agg_t, ln_g + ((size_t)l * 2 + 1) * DIM, ln_b + ((size_t)l * 2 + 1) * DIM,
            x_tweet, N_TWEET);
    }

    // outputs: concat(x_user, x_tweet) flat
    float* out = (float*)d_out;
    hipMemcpyAsync(out, x_user, (size_t)N_USER * DIM * 4,
                   hipMemcpyDeviceToDevice, stream);
    hipMemcpyAsync(out + (size_t)N_USER * DIM, x_tweet, (size_t)N_TWEET * DIM * 4,
                   hipMemcpyDeviceToDevice, stream);
    (void)out_size;
}
